// SpecializedLiftingLayer_24421184045427
// MI455X (gfx1250) — matrix operand order repaired and checked by reading, not compiled, not run on hardware
//
#include <hip/hip_runtime.h>
#include <hip/hip_bf16.h>

typedef __attribute__((ext_vector_type(16))) __bf16 v16bf;
typedef __attribute__((ext_vector_type(8)))  float  v8f;
typedef unsigned short u16;

#define PI_F      3.14159f
#define D_MODEL   2048
#define NTOK      16384          // 4 * 4096 tokens
#define WTC       944            // 112 + 464 + 368 padded concat columns
#define YSTRIDE   468            // >= 464, 8*468 % 64 == 32 -> staggered banks

// ---------------- helpers ----------------
__device__ __forceinline__ u16 f2bf(float f) {
  unsigned u = __float_as_uint(f);
  u += 0x7fffu + ((u >> 16) & 1u);          // round-to-nearest-even
  return (u16)(u >> 16);
}

union V16 { v16bf v; float4 q[2]; };

// ---------------- kernel 1: x f32 -> bf16 ----------------
__global__ void cvt_x_kernel(const float* __restrict__ x, u16* __restrict__ xbf) {
  size_t t = (size_t)blockIdx.x * blockDim.x + threadIdx.x;   // 4,194,304 threads
  size_t base = t * 8;
  float4 f0 = *(const float4*)(x + base);
  float4 f1 = *(const float4*)(x + base + 4);
  union { u16 h[8]; uint4 u; } o;
  o.h[0] = f2bf(f0.x); o.h[1] = f2bf(f0.y); o.h[2] = f2bf(f0.z); o.h[3] = f2bf(f0.w);
  o.h[4] = f2bf(f1.x); o.h[5] = f2bf(f1.y); o.h[6] = f2bf(f1.z); o.h[7] = f2bf(f1.w);
  *(uint4*)(xbf + base) = o.u;
}

// ------- kernel 2: build N-major bf16 WT[944][2048] + fused bias[944] -------
// region 0: cols [0,112)   <- W_syn (100 real)
// region 1: cols [112,576) <- W_sem (450 real)
// region 2: cols [576,944) <- W_nar (360 real)
__global__ void pack_w_kernel(const float* __restrict__ Ws, const float* __restrict__ bs,
                              const float* __restrict__ Wm, const float* __restrict__ bm,
                              const float* __restrict__ Wn, const float* __restrict__ bn,
                              u16* __restrict__ WT, float* __restrict__ bias) {
  int idx = blockIdx.x * 256 + threadIdx.x;                   // WTC*2048 threads exactly
  int j = idx >> 11, k = idx & 2047;
  float w = 0.f;
  if (j < 112)      { if (j < 100) w = Ws[(size_t)k * 100 + j]; }
  else if (j < 576) { int c = j - 112; if (c < 450) w = Wm[(size_t)k * 450 + c]; }
  else              { int c = j - 576; if (c < 360) w = Wn[(size_t)k * 360 + c]; }
  WT[idx] = f2bf(w);
  if (idx < WTC) {
    float bv = 0.f;
    if (idx < 112)      { if (idx < 100) bv = bs[idx]; }
    else if (idx < 576) { int c = idx - 112; if (c < 450) bv = bm[c]; }
    else                { int c = idx - 576; if (c < 360) bv = bn[c]; }
    bias[idx] = bv;
  }
}

// ---------------- rotor epilogue (per region, compile-time mapping) --------
template<int REGION, int HSZ, int NHEADS, int HBASE>
__device__ __forceinline__ void do_epilogue(const float* __restrict__ Y,
                                            float* __restrict__ out,
                                            int m0, int lane) {
  const int total = NHEADS * 16;
  for (int hi = lane; hi < total; hi += 32) {
    int token = hi / NHEADS;
    int head  = hi % NHEADS;
    const float* y = Y + token * YSTRIDE + head * HSZ;
    float v[HSZ]; float s2 = 0.f;
#pragma unroll
    for (int j = 0; j < HSZ; ++j) {
      float b = tanhf(y[j]) * (-0.5f * PI_F);   // B = -tanh(raw)*PI/2
      v[j] = b; s2 += b * b;
    }
    float theta = __builtin_sqrtf(s2);
    float c = __cosf(theta);
    float sinc = (theta > 1e-8f) ? (__sinf(theta) / theta) : 1.f;
    float o[32];
#pragma unroll
    for (int s = 0; s < 32; ++s) o[s] = 0.f;
    o[0] = c;
    if (REGION == 0) {                       // bivectors 6..15
#pragma unroll
      for (int j = 0; j < 10; ++j) o[6 + j] = v[j] * sinc;
    } else if (REGION == 1) {                // vectors 1..5, bivectors 6..15
#pragma unroll
      for (int j = 0; j < 5; ++j)  o[1 + j]  = v[j] * sinc;
#pragma unroll
      for (int j = 0; j < 10; ++j) o[6 + j]  = v[5 + j] * sinc;
    } else {                                 // bivectors 6..15, quadvectors 26..30
#pragma unroll
      for (int j = 0; j < 10; ++j) o[6 + j]  = v[j] * sinc;
#pragma unroll
      for (int j = 0; j < 5; ++j)  o[26 + j] = v[10 + j] * sinc;
    }
    float4* op = (float4*)(out + ((size_t)(m0 + token) * 64 + HBASE + head) * 32);
#pragma unroll
    for (int q = 0; q < 8; ++q)
      op[q] = make_float4(o[4 * q], o[4 * q + 1], o[4 * q + 2], o[4 * q + 3]);
  }
}

// ---------------- kernel 3: fused WMMA GEMM + rotor exp_map ----------------
__global__ __launch_bounds__(32)
void gemm_lift_kernel(const u16* __restrict__ Xbf, const u16* __restrict__ WT,
                      const float* __restrict__ bias, float* __restrict__ out) {
  const int r    = blockIdx.y;               // region
  const int m0   = blockIdx.x * 16;          // 16 tokens per wave
  const int lane = threadIdx.x;
  const int cb = (r == 0) ? 0 : (r == 1) ? 112 : 576;   // WT column base
  const int nt = (r == 0) ? 7 : (r == 1) ? 29 : 23;     // 16-col tiles in region

  __shared__ float Y[16 * YSTRIDE];          // 29,952 B

  const int rlo = lane & 15;
  const int hhi = lane >> 4;
  // A fragment base: row = m0+rlo, contiguous K range (lane>>4)*16..+15
  const u16* pa = Xbf + (size_t)(m0 + rlo) * D_MODEL + hhi * 16;

  auto store_tile = [&](int tile, const v8f& acc) {
    int col = tile * 16 + rlo;
    float bc = bias[cb + col];
#pragma unroll
    for (int i = 0; i < 8; ++i)
      Y[(i + 8 * hhi) * YSTRIDE + col] = acc[i] + bc;   // C: m=i+8*(lane>>4), n=lane&15
  };

  for (int tg = 0; tg < nt; tg += 4) {
    int t0 = tg;
    int t1 = min(tg + 1, nt - 1);
    int t2 = min(tg + 2, nt - 1);
    int t3 = min(tg + 3, nt - 1);
    // B fragment base: col = n0 + (lane&15), contiguous K range (lane>>4)*16..+15
    const u16* pb0 = WT + (size_t)(cb + t0 * 16 + rlo) * D_MODEL + hhi * 16;
    const u16* pb1 = WT + (size_t)(cb + t1 * 16 + rlo) * D_MODEL + hhi * 16;
    const u16* pb2 = WT + (size_t)(cb + t2 * 16 + rlo) * D_MODEL + hhi * 16;
    const u16* pb3 = WT + (size_t)(cb + t3 * 16 + rlo) * D_MODEL + hhi * 16;

    v8f a0 = {}, a1 = {}, a2 = {}, a3 = {};
#pragma unroll 2
    for (int k0 = 0; k0 < D_MODEL; k0 += 32) {
      __builtin_prefetch((const void*)(pa + k0 + 256), 0, 1);  // global_prefetch_b8
      V16 av;
      av.q[0] = *(const float4*)(pa + k0);        // K h..h+7
      av.q[1] = *(const float4*)(pa + k0 + 8);    // K h+8..h+15
      V16 b0, b1, b2, b3;
      b0.q[0] = *(const float4*)(pb0 + k0); b0.q[1] = *(const float4*)(pb0 + k0 + 8);
      b1.q[0] = *(const float4*)(pb1 + k0); b1.q[1] = *(const float4*)(pb1 + k0 + 8);
      b2.q[0] = *(const float4*)(pb2 + k0); b2.q[1] = *(const float4*)(pb2 + k0 + 8);
      b3.q[0] = *(const float4*)(pb3 + k0); b3.q[1] = *(const float4*)(pb3 + k0 + 8);
      a0 = __builtin_amdgcn_wmma_f32_16x16x32_bf16(false, av.v, false, b0.v, (short)0, a0, false, false);
      a1 = __builtin_amdgcn_wmma_f32_16x16x32_bf16(false, av.v, false, b1.v, (short)0, a1, false, false);
      a2 = __builtin_amdgcn_wmma_f32_16x16x32_bf16(false, av.v, false, b2.v, (short)0, a2, false, false);
      a3 = __builtin_amdgcn_wmma_f32_16x16x32_bf16(false, av.v, false, b3.v, (short)0, a3, false, false);
    }
    store_tile(t0, a0);
    if (t1 != t0) store_tile(t1, a1);
    if (t2 != t1) store_tile(t2, a2);
    if (t3 != t2) store_tile(t3, a3);
  }

  __syncthreads();   // single wave: DS-ordered; keeps LDS RAW safe

  if (r == 0)      do_epilogue<0, 10, 10,  0>(Y, out, m0, lane);
  else if (r == 1) do_epilogue<1, 15, 30, 10>(Y, out, m0, lane);
  else             do_epilogue<2, 15, 24, 40>(Y, out, m0, lane);
}

// ---------------- launch ----------------
extern "C" void kernel_launch(void* const* d_in, const int* in_sizes, int n_in,
                              void* d_out, int out_size, void* d_ws, size_t ws_size,
                              hipStream_t stream) {
  const float* x    = (const float*)d_in[0];
  const float* Wsyn = (const float*)d_in[1];
  const float* bsyn = (const float*)d_in[2];
  const float* Wsem = (const float*)d_in[3];
  const float* bsem = (const float*)d_in[4];
  const float* Wnar = (const float*)d_in[5];
  const float* bnar = (const float*)d_in[6];
  float* out = (float*)d_out;

  char* ws = (char*)d_ws;
  u16*   Xbf  = (u16*)ws;                                   // 16384*2048*2 = 67,108,864 B
  u16*   WT   = (u16*)(ws + 67108864);                      // 944*2048*2   =  3,866,624 B
  float* bias = (float*)(ws + 67108864 + 3866624);          // 944*4 B

  cvt_x_kernel<<<16384, 256, 0, stream>>>(x, Xbf);          // 33.5M elems, 8/thread
  pack_w_kernel<<<(WTC * 2048) / 256, 256, 0, stream>>>(Wsyn, bsyn, Wsem, bsem,
                                                        Wnar, bnar, WT, bias);
  gemm_lift_kernel<<<dim3(NTOK / 16, 3), 32, 0, stream>>>(Xbf, WT, bias, out);
}
